// BIPA_45062796869732
// MI455X (gfx1250) — matrix operand order repaired and checked by reading, not compiled, not run on hardware
//
#include <hip/hip_runtime.h>
#include <hip/hip_bf16.h>
#include <math.h>

// ---------------------------------------------------------------------------
// Types for CDNA5 WMMA (wave32, 16x16x32 bf16 -> f32)
// ---------------------------------------------------------------------------
typedef __bf16 bf16_t;
typedef __attribute__((ext_vector_type(16))) __bf16 v16bf;
typedef __attribute__((ext_vector_type(8)))  __bf16 v8bf;
typedef __attribute__((ext_vector_type(4)))  __bf16 v4bf;
typedef __attribute__((ext_vector_type(8)))  float  v8f;
typedef int v4i_vs __attribute__((vector_size(16)));   // 4 x i32, gcc-style

#define BB      32
#define IMGS    224
#define PSZ     16
#define DMODEL  768
#define NH      12
#define HDIM    64
#define LTOK    197
#define NPSD    14
#define NPATCH  196
#define MLPD    3072
#define NCLS    100
#define NLAYER  12
#define LNEPS   1e-5f

// ---------------------------------------------------------------------------
// CDNA5 async global->LDS copies (ASYNCcnt-tracked), with safe fallback.
// Builtin signatures (verified via hipcc diagnostics on this toolchain):
//   b32 : (int addrspace(1)*, int addrspace(3)*, imm offset, imm cpol)
//   b128: (v4i32 addrspace(1)*, v4i32 addrspace(3)*, imm offset, imm cpol)
// ---------------------------------------------------------------------------
#if defined(__gfx1250__)
#if __has_builtin(__builtin_amdgcn_global_load_async_to_lds_b32) && \
    __has_builtin(__builtin_amdgcn_global_load_async_to_lds_b128)
#define HAVE_ASYNC_LDS 1
#endif
#endif
#ifndef HAVE_ASYNC_LDS
#define HAVE_ASYNC_LDS 0
#endif

__device__ __forceinline__ void g2l_async_b32(const void* g, void* l) {
#if HAVE_ASYNC_LDS
  __builtin_amdgcn_global_load_async_to_lds_b32(
      (__attribute__((address_space(1))) int*)((void*)g),
      (__attribute__((address_space(3))) int*)l, 0, 0);
#else
  *(float*)l = *(const float*)g;
#endif
}

__device__ __forceinline__ void g2l_async_b128(const void* g, void* l) {
#if HAVE_ASYNC_LDS
  __builtin_amdgcn_global_load_async_to_lds_b128(
      (__attribute__((address_space(1))) v4i_vs*)((void*)g),
      (__attribute__((address_space(3))) v4i_vs*)l, 0, 0);
#else
  *(float4*)l = *(const float4*)g;
#endif
}

__device__ __forceinline__ void async_wait_all() {
#if HAVE_ASYNC_LDS
  __builtin_amdgcn_s_wait_asynccnt(0);
#endif
}

// ---------------------------------------------------------------------------
// WMMA helpers
// ---------------------------------------------------------------------------
__device__ __forceinline__ v8f wmma_bf16(v16bf a, v16bf b, v8f c) {
#if defined(__gfx1250__)
  return __builtin_amdgcn_wmma_f32_16x16x32_bf16(false, a, false, b, (short)0, c,
                                                 false, false);
#else
  return c;
#endif
}

__device__ __forceinline__ v16bf cat8(v8bf lo, v8bf hi) {
  v16bf r;
#pragma unroll
  for (int e = 0; e < 8; ++e) { r[e] = lo[e]; r[e + 8] = hi[e]; }
  return r;
}

// A fragment: per-lane 8 contiguous bf16 at p and 8 at p+16
__device__ __forceinline__ v16bf ldfragA(const bf16_t* p) {
  return cat8(*(const v8bf*)p, *(const v8bf*)(p + 16));
}
// B fragment: per-lane 8 contiguous bf16 at p and 8 at p+16
__device__ __forceinline__ v16bf ldfragB(const bf16_t* p) {
  return cat8(*(const v8bf*)p, *(const v8bf*)(p + 16));
}

// ---------------------------------------------------------------------------
// fp32 -> bf16 convert (RNE), vectorized 4 elems/thread
// ---------------------------------------------------------------------------
__global__ void k_f32_to_bf16(const float* __restrict__ in,
                              bf16_t* __restrict__ out, int n) {
  int i = (blockIdx.x * blockDim.x + threadIdx.x) * 4;
  if (i + 3 < n) {
    float4 f = *(const float4*)(in + i);
    v4bf o;
    o[0] = (bf16_t)f.x; o[1] = (bf16_t)f.y; o[2] = (bf16_t)f.z; o[3] = (bf16_t)f.w;
    *(v4bf*)(out + i) = o;
  } else {
    for (; i < n; ++i) out[i] = (bf16_t)in[i];
  }
}

// ---------------------------------------------------------------------------
// im2col for patch embed: A[b*196+p][c*256+py*16+px] = x[b,c,...]  (bf16)
// ---------------------------------------------------------------------------
__global__ void k_build_patches(const float* __restrict__ x,
                                bf16_t* __restrict__ patches) {
  int i = blockIdx.x * blockDim.x + threadIdx.x;
  if (i >= BB * NPATCH * DMODEL) return;
  int col = i % DMODEL;
  int row = i / DMODEL;
  int b = row / NPATCH, p = row % NPATCH;
  int c = col >> 8, rem = col & 255;
  int py = rem >> 4, px = rem & 15;
  int iy = (p / NPSD) * PSZ + py;
  int ix = (p % NPSD) * PSZ + px;
  patches[i] = (bf16_t)x[((size_t)(b * 3 + c) * IMGS + iy) * IMGS + ix];
}

// ---------------------------------------------------------------------------
// cls token row of t + zero the cls column of mbc
// ---------------------------------------------------------------------------
__global__ void k_init_cls(const float* __restrict__ cls,
                           const float* __restrict__ pos,
                           float* __restrict__ t, float* __restrict__ mbc) {
  int i = blockIdx.x * blockDim.x + threadIdx.x;
  if (i >= BB * DMODEL) return;
  int b = i / DMODEL, d = i % DMODEL;
  t[(size_t)b * LTOK * DMODEL + d] = cls[d] + pos[d];
  if (d == 0) mbc[b * LTOK] = 0.f;
}

// ---------------------------------------------------------------------------
// WMMA GEMM: C[M,N] = A[M,K] * Bt[N,K]^T (+bias, fused epilogues)
//   block = 256 threads = 8 waves; wave -> 32x64 tile; block -> 256x64
//   Branch-free K-loop (rows clamped on load) + software double-buffering.
//   __launch_bounds__(256, 1): relax the VGPR budget (~180 live regs) so the
//   double-buffered fragments are NOT spilled to scratch.
//   mode 0: outf[r*N+c] = acc + bias
//   mode 1: outb[r*N+c] = bf16(gelu_exact(acc + bias))
//   mode 2: outf[r*N+c] += acc + bias                 (residual add into t)
//   mode 3: patch embed: t[(b*197+1+p)*768+c] = acc + bias + pos[(1+p)*768+c]
// ---------------------------------------------------------------------------
__global__ void __launch_bounds__(256, 1)
k_gemm_wmma(const bf16_t* __restrict__ A,
            const bf16_t* __restrict__ Bt,
            const float* __restrict__ bias,
            int M, int N, int K, int mode,
            float* __restrict__ outf,
            bf16_t* __restrict__ outb,
            const float* __restrict__ extra) {
  const int lane = threadIdx.x & 31;
  const int wave = threadIdx.x >> 5;
  const int n0 = blockIdx.x * 64;
  const int r0 = blockIdx.y * 256 + wave * 32;
  const int mrow = lane & 15;
  const int khiA = (lane >> 4) * 8;   // A: K-half offset per lane group
  const int khiB = (lane >> 4) * 8;   // B: K-half offset per lane group

  v8f acc[2][4];
#pragma unroll
  for (int i = 0; i < 2; ++i)
#pragma unroll
    for (int j = 0; j < 4; ++j)
#pragma unroll
      for (int e = 0; e < 8; ++e) acc[i][j][e] = 0.f;

  // clamp rows so loads are always in-bounds and unconditional
  int ra0 = r0 + mrow;       if (ra0 > M - 1) ra0 = M - 1;
  int ra1 = r0 + 16 + mrow;  if (ra1 > M - 1) ra1 = M - 1;
  const bf16_t* Arow0 = A + (size_t)ra0 * K + khiA;
  const bf16_t* Arow1 = A + (size_t)ra1 * K + khiA;
  const bf16_t* Brow0 = Bt + (size_t)(n0 + mrow) * K + khiB;
  const bf16_t* Brow1 = Brow0 + (size_t)16 * K;
  const bf16_t* Brow2 = Brow0 + (size_t)32 * K;
  const bf16_t* Brow3 = Brow0 + (size_t)48 * K;

  // prologue: fragments for k=0
  v16bf a0 = ldfragA(Arow0), a1 = ldfragA(Arow1);
  v16bf b0 = ldfragB(Brow0), b1 = ldfragB(Brow1);
  v16bf b2 = ldfragB(Brow2), b3 = ldfragB(Brow3);

  for (int k0 = 32; k0 < K; k0 += 32) {
    // prefetch next K-step while current WMMAs execute
    v16bf na0 = ldfragA(Arow0 + k0), na1 = ldfragA(Arow1 + k0);
    v16bf nb0 = ldfragB(Brow0 + k0), nb1 = ldfragB(Brow1 + k0);
    v16bf nb2 = ldfragB(Brow2 + k0), nb3 = ldfragB(Brow3 + k0);

    acc[0][0] = wmma_bf16(a0, b0, acc[0][0]);
    acc[1][0] = wmma_bf16(a1, b0, acc[1][0]);
    acc[0][1] = wmma_bf16(a0, b1, acc[0][1]);
    acc[1][1] = wmma_bf16(a1, b1, acc[1][1]);
    acc[0][2] = wmma_bf16(a0, b2, acc[0][2]);
    acc[1][2] = wmma_bf16(a1, b2, acc[1][2]);
    acc[0][3] = wmma_bf16(a0, b3, acc[0][3]);
    acc[1][3] = wmma_bf16(a1, b3, acc[1][3]);

    a0 = na0; a1 = na1; b0 = nb0; b1 = nb1; b2 = nb2; b3 = nb3;
  }
  // tail K-step
  acc[0][0] = wmma_bf16(a0, b0, acc[0][0]);
  acc[1][0] = wmma_bf16(a1, b0, acc[1][0]);
  acc[0][1] = wmma_bf16(a0, b1, acc[0][1]);
  acc[1][1] = wmma_bf16(a1, b1, acc[1][1]);
  acc[0][2] = wmma_bf16(a0, b2, acc[0][2]);
  acc[1][2] = wmma_bf16(a1, b2, acc[1][2]);
  acc[0][3] = wmma_bf16(a0, b3, acc[0][3]);
  acc[1][3] = wmma_bf16(a1, b3, acc[1][3]);

  const int ncol  = n0 + (lane & 15);
  const int rbase = r0 + ((lane >> 4) * 8);
#pragma unroll
  for (int i = 0; i < 2; ++i) {
#pragma unroll
    for (int j = 0; j < 4; ++j) {
      const int gc = ncol + j * 16;
      const float bv = bias ? bias[gc] : 0.f;
#pragma unroll
      for (int r = 0; r < 8; ++r) {
        const int gr = rbase + i * 16 + r;
        if (gr >= M) continue;
        float v = acc[i][j][r] + bv;
        if (mode == 0) {
          outf[(size_t)gr * N + gc] = v;
        } else if (mode == 1) {
          float g = 0.5f * v * (1.f + erff(v * 0.70710678118f));
          outb[(size_t)gr * N + gc] = (bf16_t)g;
        } else if (mode == 2) {
          outf[(size_t)gr * N + gc] += v;
        } else {  // mode 3: patch-embed epilogue
          int b = gr / NPATCH, p = gr % NPATCH;
          outf[((size_t)b * LTOK + 1 + p) * DMODEL + gc] =
              v + extra[(size_t)(1 + p) * DMODEL + gc];
        }
      }
    }
  }
}

// ---------------------------------------------------------------------------
// LayerNorm over D=768; one block (256 thr) per row.
//   row_mul: stride multiplier (197 for cls-only final norm)
//   writes bf16 (outb) or fp32 (outf) compact [row][768]
// ---------------------------------------------------------------------------
__global__ void k_layernorm(const float* __restrict__ in,
                            const float* __restrict__ g,
                            const float* __restrict__ bta,
                            bf16_t* __restrict__ outb,
                            float* __restrict__ outf, int row_mul) {
  __shared__ float red[256];
  const int row = blockIdx.x;
  const int tid = threadIdx.x;
  const float* src = in + (size_t)row * row_mul * DMODEL;
  float x0 = src[tid], x1 = src[tid + 256], x2 = src[tid + 512];
  red[tid] = x0 + x1 + x2;
  __syncthreads();
  for (int s = 128; s > 0; s >>= 1) {
    if (tid < s) red[tid] += red[tid + s];
    __syncthreads();
  }
  float mean = red[0] * (1.f / DMODEL);
  __syncthreads();
  float d0 = x0 - mean, d1 = x1 - mean, d2 = x2 - mean;
  red[tid] = d0 * d0 + d1 * d1 + d2 * d2;
  __syncthreads();
  for (int s = 128; s > 0; s >>= 1) {
    if (tid < s) red[tid] += red[tid + s];
    __syncthreads();
  }
  float rs = rsqrtf(red[0] * (1.f / DMODEL) + LNEPS);
  float y0 = d0 * rs * g[tid]       + bta[tid];
  float y1 = d1 * rs * g[tid + 256] + bta[tid + 256];
  float y2 = d2 * rs * g[tid + 512] + bta[tid + 512];
  size_t o = (size_t)row * DMODEL + tid;
  if (outb) {
    outb[o] = (bf16_t)y0; outb[o + 256] = (bf16_t)y1; outb[o + 512] = (bf16_t)y2;
  } else {
    outf[o] = y0; outf[o + 256] = y1; outf[o + 512] = y2;
  }
}

// ---------------------------------------------------------------------------
// Attention: one block (256 thr) per (b,h). K,V staged into LDS once via
// CDNA5 async global->LDS loads (ASYNCcnt); per-row softmax with alpha*mbc
// key bias fused; writes o in bf16 for the proj GEMM.
// ---------------------------------------------------------------------------
__global__ void k_attn(const float* __restrict__ qkv,
                       const float* __restrict__ mbc,
                       const float* __restrict__ alpha, int layer,
                       bf16_t* __restrict__ o_bf) {
  extern __shared__ float sm[];
  float* Ksh  = sm;                 // 197*65
  float* Vsh  = Ksh + 197 * 65;     // 197*65
  float* qsh  = Vsh + 197 * 65;     // 64
  float* lg   = qsh + 64;           // 256
  float* red  = lg + 256;           // 256
  float* part = red + 256;          // 256

  const int tid = threadIdx.x;
  const int bh = blockIdx.x;
  const int b = bh / NH, h = bh % NH;
  const float a = alpha[layer];
  const float scale = 0.125f;  // 64^-0.5

  for (int idx = tid; idx < LTOK * HDIM; idx += 256) {
    int m = idx >> 6, d = idx & 63;
    const float* base = qkv + (size_t)(b * LTOK + m) * 2304 + h * 64 + d;
    g2l_async_b32(base + 768,  &Ksh[m * 65 + d]);
    g2l_async_b32(base + 1536, &Vsh[m * 65 + d]);
  }
  async_wait_all();
  __syncthreads();

  for (int l = 0; l < LTOK; ++l) {
    if (tid < 64)
      qsh[tid] = qkv[(size_t)(b * LTOK + l) * 2304 + h * 64 + tid];
    __syncthreads();

    float e = -3.0e38f;
    if (tid < LTOK) {
      float dot = 0.f;
      for (int d = 0; d < 64; ++d) dot += qsh[d] * Ksh[tid * 65 + d];
      e = dot * scale + a * mbc[b * LTOK + tid];
    }
    red[tid] = e;
    __syncthreads();
    for (int s = 128; s > 0; s >>= 1) {
      if (tid < s) red[tid] = fmaxf(red[tid], red[tid + s]);
      __syncthreads();
    }
    float mx = red[0];
    __syncthreads();
    float ex = (tid < LTOK) ? __expf(e - mx) : 0.f;
    lg[tid] = ex;
    red[tid] = ex;
    __syncthreads();
    for (int s = 128; s > 0; s >>= 1) {
      if (tid < s) red[tid] += red[tid + s];
      __syncthreads();
    }
    float inv = 1.f / red[0];

    const int d = tid & 63, seg = tid >> 6;
    float av = 0.f;
    for (int m = seg; m < LTOK; m += 4) av += lg[m] * Vsh[m * 65 + d];
    part[tid] = av;
    __syncthreads();
    if (tid < 64) {
      float o = (part[tid] + part[tid + 64] + part[tid + 128] + part[tid + 192]) * inv;
      o_bf[(size_t)(b * LTOK + l) * DMODEL + h * 64 + tid] = (bf16_t)o;
    }
    __syncthreads();
  }
}

// ---------------------------------------------------------------------------
// Fused BAM CNN: one block per (b, patch). Computes conv1->bn/relu ->
// conv2->bn/relu -> conv3->bn/relu -> 1x1 out for one 16x16 patch tile
// (receptive field 22x22) entirely in LDS, then mean -> sigmoid -> mb/mbc.
// Conv weights staged into LDS with async b128 copies.
// ---------------------------------------------------------------------------
__global__ void k_bam(const float* __restrict__ x,
                      const float* __restrict__ c1w, const float* __restrict__ c1b,
                      const float* __restrict__ bn1g, const float* __restrict__ bn1b,
                      const float* __restrict__ bn1m, const float* __restrict__ bn1v,
                      const float* __restrict__ c2w, const float* __restrict__ c2b,
                      const float* __restrict__ bn2g, const float* __restrict__ bn2b,
                      const float* __restrict__ bn2m, const float* __restrict__ bn2v,
                      const float* __restrict__ c3w, const float* __restrict__ c3b,
                      const float* __restrict__ bn3g, const float* __restrict__ bn3b,
                      const float* __restrict__ bn3m, const float* __restrict__ bn3v,
                      const float* __restrict__ outw, const float* __restrict__ outb,
                      float* __restrict__ raw_out, float* __restrict__ mb_out,
                      float* __restrict__ mbc) {
  extern __shared__ float sm[];
  float* sx  = sm;            // 3*22*22   = 1452
  float* s1  = sx + 1452;     // 32*20*20  = 12800
  float* s2  = s1 + 12800;    // 32*18*18  = 10368
  float* s3  = s2 + 10368;    // 16*16*16  = 4096
  float* w1  = s3 + 4096;     // 864   (16B-aligned offset)
  float* w2  = w1 + 864;      // 9216
  float* w3  = w2 + 9216;     // 4608
  float* bns = w3 + 4608;     // 160
  float* red = bns + 160;     // 256

  const int tid = threadIdx.x;
  const int px = blockIdx.x, py = blockIdx.y, b = blockIdx.z;
  const int oy = py * 16, ox = px * 16;

  // async-stage conv weights (sizes are multiples of 4 floats, 16B aligned)
  for (int i = tid; i < 216;  i += 256) g2l_async_b128(c1w + 4 * i, w1 + 4 * i);
  for (int i = tid; i < 2304; i += 256) g2l_async_b128(c2w + 4 * i, w2 + 4 * i);
  for (int i = tid; i < 1152; i += 256) g2l_async_b128(c3w + 4 * i, w3 + 4 * i);
  if (tid < 32) {
    float s = bn1g[tid] * rsqrtf(bn1v[tid] + LNEPS);
    bns[tid] = s; bns[32 + tid] = bn1b[tid] - bn1m[tid] * s;
  } else if (tid < 64) {
    int c = tid - 32;
    float s = bn2g[c] * rsqrtf(bn2v[c] + LNEPS);
    bns[64 + c] = s; bns[96 + c] = bn2b[c] - bn2m[c] * s;
  } else if (tid < 80) {
    int c = tid - 64;
    float s = bn3g[c] * rsqrtf(bn3v[c] + LNEPS);
    bns[128 + c] = s; bns[144 + c] = bn3b[c] - bn3m[c] * s;
  }
  for (int i = tid; i < 3 * 22 * 22; i += 256) {
    int c = i / 484, rem = i % 484;
    int r = rem / 22, cc = rem % 22;
    int gy = oy - 3 + r, gx = ox - 3 + cc;
    float v = 0.f;
    if (gy >= 0 && gy < IMGS && gx >= 0 && gx < IMGS)
      v = x[((size_t)(b * 3 + c) * IMGS + gy) * IMGS + gx];
    sx[i] = v;
  }
  async_wait_all();
  __syncthreads();

  // conv1: 3->32, output 20x20 region
  for (int i = tid; i < 32 * 400; i += 256) {
    int co = i / 400, rem = i % 400;
    int r = rem / 20, cc = rem % 20;
    float acc = c1b[co];
    for (int ci = 0; ci < 3; ++ci) {
      const float* srow = &sx[ci * 484 + r * 22 + cc];
      const float* wr = &w1[(co * 3 + ci) * 9];
      acc += srow[0] * wr[0] + srow[1] * wr[1] + srow[2] * wr[2]
           + srow[22] * wr[3] + srow[23] * wr[4] + srow[24] * wr[5]
           + srow[44] * wr[6] + srow[45] * wr[7] + srow[46] * wr[8];
    }
    float v = acc * bns[co] + bns[32 + co];
    s1[i] = v > 0.f ? v : 0.f;
  }
  __syncthreads();

  // conv2: 32->32, output 18x18 region
  for (int i = tid; i < 32 * 324; i += 256) {
    int co = i / 324, rem = i % 324;
    int r = rem / 18, cc = rem % 18;
    float acc = c2b[co];
    for (int ci = 0; ci < 32; ++ci) {
      const float* srow = &s1[ci * 400 + r * 20 + cc];
      const float* wr = &w2[(co * 32 + ci) * 9];
      acc += srow[0] * wr[0] + srow[1] * wr[1] + srow[2] * wr[2]
           + srow[20] * wr[3] + srow[21] * wr[4] + srow[22] * wr[5]
           + srow[40] * wr[6] + srow[41] * wr[7] + srow[42] * wr[8];
    }
    float v = acc * bns[64 + co] + bns[96 + co];
    s2[i] = v > 0.f ? v : 0.f;
  }
  __syncthreads();

  // conv3: 32->16, output 16x16 region
  for (int i = tid; i < 16 * 256; i += 256) {
    int co = i / 256, rem = i % 256;
    int r = rem / 16, cc = rem % 16;
    float acc = c3b[co];
    for (int ci = 0; ci < 32; ++ci) {
      const float* srow = &s2[ci * 324 + r * 18 + cc];
      const float* wr = &w3[(co * 32 + ci) * 9];
      acc += srow[0] * wr[0] + srow[1] * wr[1] + srow[2] * wr[2]
           + srow[18] * wr[3] + srow[19] * wr[4] + srow[20] * wr[5]
           + srow[36] * wr[6] + srow[37] * wr[7] + srow[38] * wr[8];
    }
    float v = acc * bns[128 + co] + bns[144 + co];
    s3[i] = v > 0.f ? v : 0.f;
  }
  __syncthreads();

  // 1x1 out conv + patch mean + sigmoid
  {
    int r = tid >> 4, cc = tid & 15;
    float acc = outb[0];
    for (int ci = 0; ci < 16; ++ci) acc += s3[ci * 256 + r * 16 + cc] * outw[ci];
    raw_out[((size_t)b * IMGS + oy + r) * IMGS + ox + cc] = acc;
    red[tid] = acc;
  }
  __syncthreads();
  for (int s = 128; s > 0; s >>= 1) {
    if (tid < s) red[tid] += red[tid + s];
    __syncthreads();
  }
  if (tid == 0) {
    float mean = red[0] * (1.f / 256.f);
    float sig = 1.f / (1.f + __expf(-mean));
    int p = py * NPSD + px;
    mb_out[b * NPATCH + p] = sig;
    mbc[b * LTOK + 1 + p] = sig;
  }
}

// ---------------------------------------------------------------------------
// Classification head: logits[b,c] = clsn[b,:] . head_w[c,:] + head_b[c]
// ---------------------------------------------------------------------------
__global__ void k_head(const float* __restrict__ clsn,
                       const float* __restrict__ hw,
                       const float* __restrict__ hb,
                       float* __restrict__ logits) {
  int i = blockIdx.x * blockDim.x + threadIdx.x;
  if (i >= BB * NCLS) return;
  int b = i / NCLS, c = i % NCLS;
  const float* xr = clsn + (size_t)b * DMODEL;
  const float* wr = hw + (size_t)c * DMODEL;
  float s = hb[c];
  for (int d = 0; d < DMODEL; ++d) s += xr[d] * wr[d];
  logits[i] = s;
}

// ---------------------------------------------------------------------------
// Host orchestration
// ---------------------------------------------------------------------------
extern "C" void kernel_launch(void* const* d_in, const int* in_sizes, int n_in,
                              void* d_out, int out_size, void* d_ws, size_t ws_size,
                              hipStream_t stream) {
  (void)in_sizes; (void)n_in; (void)out_size; (void)ws_size;
  const float* x        = (const float*)d_in[0];
  const float* patch_w  = (const float*)d_in[1];
  const float* patch_b  = (const float*)d_in[2];
  const float* cls_tok  = (const float*)d_in[3];
  const float* pos_emb  = (const float*)d_in[4];
  const float* c1w = (const float*)d_in[5];  const float* c1b = (const float*)d_in[6];
  const float* bn1g = (const float*)d_in[7]; const float* bn1b = (const float*)d_in[8];
  const float* bn1m = (const float*)d_in[9]; const float* bn1v = (const float*)d_in[10];
  const float* c2w = (const float*)d_in[11]; const float* c2b = (const float*)d_in[12];
  const float* bn2g = (const float*)d_in[13]; const float* bn2b = (const float*)d_in[14];
  const float* bn2m = (const float*)d_in[15]; const float* bn2v = (const float*)d_in[16];
  const float* c3w = (const float*)d_in[17]; const float* c3b = (const float*)d_in[18];
  const float* bn3g = (const float*)d_in[19]; const float* bn3b = (const float*)d_in[20];
  const float* bn3m = (const float*)d_in[21]; const float* bn3v = (const float*)d_in[22];
  const float* outw = (const float*)d_in[23]; const float* outcb = (const float*)d_in[24];
  const float* ln1_g = (const float*)d_in[25]; const float* ln1_b = (const float*)d_in[26];
  const float* qkv_w = (const float*)d_in[27]; const float* qkv_b = (const float*)d_in[28];
  const float* alpha = (const float*)d_in[29];
  const float* proj_w = (const float*)d_in[30]; const float* proj_b = (const float*)d_in[31];
  const float* ln2_g = (const float*)d_in[32]; const float* ln2_b = (const float*)d_in[33];
  const float* mlp_w1 = (const float*)d_in[34]; const float* mlp_b1 = (const float*)d_in[35];
  const float* mlp_w2 = (const float*)d_in[36]; const float* mlp_b2 = (const float*)d_in[37];
  const float* norm_g = (const float*)d_in[38]; const float* norm_b = (const float*)d_in[39];
  const float* head_w = (const float*)d_in[40]; const float* head_b = (const float*)d_in[41];

  const int M = BB * LTOK;  // 6304 token rows

  // ---- workspace carving ----
  char* ws = (char*)d_ws;
  size_t off = 0;
  auto carve = [&](size_t bytes) -> char* {
    char* p = ws + off;
    off += (bytes + 255) & ~(size_t)255;
    return p;
  };
  float*  t       = (float*) carve((size_t)M * DMODEL * 4);
  float*  mbc     = (float*) carve((size_t)M * 4);
  bf16_t* patches = (bf16_t*)carve((size_t)BB * NPATCH * DMODEL * 2);
  bf16_t* y_bf    = (bf16_t*)carve((size_t)M * DMODEL * 2);
  float*  qkv     = (float*) carve((size_t)M * 3 * DMODEL * 4);
  bf16_t* o_bf    = (bf16_t*)carve((size_t)M * DMODEL * 2);
  bf16_t* h1_bf   = (bf16_t*)carve((size_t)M * MLPD * 2);
  bf16_t* wbuf    = (bf16_t*)carve((size_t)MLPD * DMODEL * 2);
  bf16_t* pw_bf   = (bf16_t*)carve((size_t)DMODEL * DMODEL * 2);
  float*  clsn    = (float*) carve((size_t)BB * DMODEL * 4);

  float* out_logits = (float*)d_out;
  float* out_raw    = out_logits + BB * NCLS;
  float* out_mb     = out_raw + (size_t)BB * IMGS * IMGS;

  const size_t attn_smem = (size_t)(2 * 197 * 65 + 64 + 256 + 256 + 256) * 4;
  const size_t bam_smem  = (size_t)(1452 + 12800 + 10368 + 4096 + 864 + 9216 + 4608 + 160 + 256) * 4;
  (void)hipFuncSetAttribute(reinterpret_cast<const void*>(&k_attn),
                            hipFuncAttributeMaxDynamicSharedMemorySize, (int)attn_smem);
  (void)hipFuncSetAttribute(reinterpret_cast<const void*>(&k_bam),
                            hipFuncAttributeMaxDynamicSharedMemorySize, (int)bam_smem);

  auto cdiv = [](int a, int b) { return (a + b - 1) / b; };

  // ---- patch-embed inputs + cls/pos init ----
  {
    int n = DMODEL * DMODEL;
    k_f32_to_bf16<<<cdiv(n, 1024), 256, 0, stream>>>(patch_w, pw_bf, n);
  }
  {
    int n = BB * NPATCH * DMODEL;
    k_build_patches<<<cdiv(n, 256), 256, 0, stream>>>(x, patches);
  }
  k_init_cls<<<cdiv(BB * DMODEL, 256), 256, 0, stream>>>(cls_tok, pos_emb, t, mbc);

  // ---- fused BAM CNN -> raw_bam, mb, mbc ----
  k_bam<<<dim3(NPSD, NPSD, BB), 256, bam_smem, stream>>>(
      x, c1w, c1b, bn1g, bn1b, bn1m, bn1v,
      c2w, c2b, bn2g, bn2b, bn2m, bn2v,
      c3w, c3b, bn3g, bn3b, bn3m, bn3v,
      outw, outcb, out_raw, out_mb, mbc);

  // ---- patch embed GEMM: t[b,1+p,:] = patches @ patch_w^T + b + pos ----
  k_gemm_wmma<<<dim3(DMODEL / 64, cdiv(BB * NPATCH, 256)), 256, 0, stream>>>(
      patches, pw_bf, patch_b, BB * NPATCH, DMODEL, DMODEL, 3,
      t, nullptr, pos_emb);

  // ---- transformer layers ----
  for (int l = 0; l < NLAYER; ++l) {
    // LN1 -> y_bf
    k_layernorm<<<M, 256, 0, stream>>>(t, ln1_g + l * DMODEL, ln1_b + l * DMODEL,
                                       y_bf, nullptr, 1);
    // qkv GEMM
    {
      int n = 3 * DMODEL * DMODEL;
      k_f32_to_bf16<<<cdiv(n, 1024), 256, 0, stream>>>(qkv_w + (size_t)l * n, wbuf, n);
    }
    k_gemm_wmma<<<dim3(3 * DMODEL / 64, cdiv(M, 256)), 256, 0, stream>>>(
        y_bf, wbuf, qkv_b + l * 3 * DMODEL, M, 3 * DMODEL, DMODEL, 0,
        qkv, nullptr, nullptr);
    // attention (flash-style, K/V in LDS, BAM key-bias fused)
    k_attn<<<BB * NH, 256, attn_smem, stream>>>(qkv, mbc, alpha, l, o_bf);
    // proj GEMM, residual add into t
    {
      int n = DMODEL * DMODEL;
      k_f32_to_bf16<<<cdiv(n, 1024), 256, 0, stream>>>(proj_w + (size_t)l * n, wbuf, n);
    }
    k_gemm_wmma<<<dim3(DMODEL / 64, cdiv(M, 256)), 256, 0, stream>>>(
        o_bf, wbuf, proj_b + l * DMODEL, M, DMODEL, DMODEL, 2,
        t, nullptr, nullptr);
    // LN2 -> y_bf
    k_layernorm<<<M, 256, 0, stream>>>(t, ln2_g + l * DMODEL, ln2_b + l * DMODEL,
                                       y_bf, nullptr, 1);
    // MLP1 GEMM with fused exact-GELU -> bf16
    {
      int n = MLPD * DMODEL;
      k_f32_to_bf16<<<cdiv(n, 1024), 256, 0, stream>>>(mlp_w1 + (size_t)l * n, wbuf, n);
    }
    k_gemm_wmma<<<dim3(MLPD / 64, cdiv(M, 256)), 256, 0, stream>>>(
        y_bf, wbuf, mlp_b1 + l * MLPD, M, MLPD, DMODEL, 1,
        nullptr, h1_bf, nullptr);
    // MLP2 GEMM, residual add into t
    {
      int n = DMODEL * MLPD;
      k_f32_to_bf16<<<cdiv(n, 1024), 256, 0, stream>>>(mlp_w2 + (size_t)l * n, wbuf, n);
    }
    k_gemm_wmma<<<dim3(DMODEL / 64, cdiv(M, 256)), 256, 0, stream>>>(
        h1_bf, wbuf, mlp_b2 + l * DMODEL, M, DMODEL, MLPD, 2,
        t, nullptr, nullptr);
  }

  // ---- final norm on cls rows + head ----
  k_layernorm<<<BB, 256, 0, stream>>>(t, norm_g, norm_b, nullptr, clsn, LTOK);
  k_head<<<cdiv(BB * NCLS, 256), 256, 0, stream>>>(clsn, head_w, head_b, out_logits);
}
